// MultiViewGAT_69793218560248
// MI455X (gfx1250) — matrix operand order repaired and checked by reading, not compiled, not run on hardware
//
#include <hip/hip_runtime.h>

// ---------- types ----------
typedef __attribute__((ext_vector_type(16))) __bf16 v16bf;
typedef __attribute__((ext_vector_type(8)))  float  v8f;
typedef __attribute__((ext_vector_type(4)))  int    v4i;
typedef __attribute__((ext_vector_type(4)))  float  v4f;

union ABPack { v16bf bf; unsigned short us[16]; v4i q[2]; };

constexpr int   NN    = 6144;   // nodes
constexpr int   IND   = 512;    // input dim
constexpr int   DD    = 256;    // hidden dim
constexpr int   OUTD  = 128;    // out dim
constexpr float ALPHA_ = 0.2f;
constexpr float EPS_   = 1e-5f;

// ---------- workspace layout (bytes) : total ~25.5 MB ----------
constexpr size_t OFF_XBF   = 0;                         // ushort [6144][512]
constexpr size_t OFF_WT    = OFF_XBF  + 6291456;        // ushort [256][512]
constexpr size_t OFF_H     = OFF_WT   + 262144;         // float  [6144][256]
constexpr size_t OFF_HT    = OFF_H    + 6291456;        // ushort [256][6144]
constexpr size_t OFF_OUTV  = OFF_HT   + 3145728;        // float  [6144][256]
constexpr size_t OFF_ZACC  = OFF_OUTV + 6291456;        // float  [6144][128]
constexpr size_t OFF_S1    = OFF_ZACC + 3145728;        // float  [6144]
constexpr size_t OFF_S2    = OFF_S1   + 24576;          // float  [6144]
constexpr size_t OFF_S2MAX = OFF_S2   + 24576;          // float  [4]
constexpr size_t OFF_WVEC  = OFF_S2MAX + 16;            // float  [4]

__device__ inline unsigned short f2bf(float f) {
  unsigned u = __float_as_uint(f);
  u += 0x7FFFu + ((u >> 16) & 1u);          // round-to-nearest-even
  return (unsigned short)(u >> 16);
}

// ---------- tiny kernels ----------
__global__ void k_softmax_alpha(const float* __restrict__ a,
                                float* __restrict__ wvec,
                                float* __restrict__ wout) {
  if (threadIdx.x == 0) {
    float a0 = a[0], a1 = a[1], a2 = a[2];
    float m  = fmaxf(a0, fmaxf(a1, a2));
    float e0 = __expf(a0 - m), e1 = __expf(a1 - m), e2 = __expf(a2 - m);
    float inv = 1.f / (e0 + e1 + e2);
    wvec[0] = e0 * inv; wvec[1] = e1 * inv; wvec[2] = e2 * inv;
    wout[0] = wvec[0];  wout[1] = wvec[1];  wout[2] = wvec[2];
  }
}

__global__ void k_convert_x(const float* __restrict__ x,
                            unsigned short* __restrict__ xbf, int n) {
  int i = blockIdx.x * blockDim.x + threadIdx.x;
  if (i < n) xbf[i] = f2bf(x[i]);
}

__global__ void k_transpose_w(const float* __restrict__ W,          // [512][256]
                              unsigned short* __restrict__ wt) {    // [256][512]
  int i = blockIdx.x * blockDim.x + threadIdx.x;
  if (i < IND * DD) {
    int k = i >> 8;        // row in W   (0..511)
    int n = i & 255;       // col in W   (0..255)
    wt[n * IND + k] = f2bf(W[i]);
  }
}

// ---------- GEMM: h = x @ W  (bf16 WMMA, f32 acc) ----------
// grid = 1536 blocks x 128 threads; each wave computes one 16x16 tile.
__global__ __launch_bounds__(128)
void k_gemm_h(const unsigned short* __restrict__ xbf,   // [6144][512] bf16
              const unsigned short* __restrict__ wt,    // [256][512]  bf16 (W^T)
              float* __restrict__ h,                    // [6144][256]
              unsigned short* __restrict__ ht) {        // [256][6144] bf16
  const int lane = threadIdx.x & 31;
  const int wid  = threadIdx.x >> 5;
  const int gw   = blockIdx.x * 4 + wid;
  const int mt   = gw >> 4;               // 0..383
  const int nt   = gw & 15;               // 0..15
  const int l15  = lane & 15;
  const int hi8  = (lane >> 4) << 3;      // 0 or 8
  const int hi16 = (lane >> 4) << 4;      // 0 or 16
  const unsigned short* ap = xbf + (mt * 16 + l15) * IND;
  const unsigned short* bp = wt  + (nt * 16 + l15) * IND;
  v8f c = {0.f, 0.f, 0.f, 0.f, 0.f, 0.f, 0.f, 0.f};
#pragma unroll 4
  for (int kc = 0; kc < IND; kc += 32) {
    ABPack a, b;
    // A (16x32, interleaved K-pair layout): runs [hi8..hi8+7] and [16+hi8..23+hi8]
    a.q[0] = *(const v4i*)(ap + kc + hi8);
    a.q[1] = *(const v4i*)(ap + kc + 16 + hi8);
    // B (32x16, same interleaved layout as A): runs [hi8..hi8+7] and [16+hi8..23+hi8]
    b.q[0] = *(const v4i*)(bp + kc + hi8);
    b.q[1] = *(const v4i*)(bp + kc + 16 + hi8);
    c = __builtin_amdgcn_wmma_f32_16x16x32_bf16(false, a.bf, false, b.bf,
                                                (short)0, c, false, false);
  }
#pragma unroll
  for (int v = 0; v < 8; ++v) {
    int r   = mt * 16 + v + hi8;          // C/D layout: row = vgpr + 8*(lane>=16)
    int col = nt * 16 + l15;
    float val = c[v];
    h[r * DD + col]  = val;
    ht[col * NN + r] = f2bf(val);
  }
}

// ---------- attention scores s1 = h.a_src, s2 = h.a_dst ----------
__global__ __launch_bounds__(128)
void k_scores(const float* __restrict__ h, const float* __restrict__ as,
              const float* __restrict__ ad, float* __restrict__ s1,
              float* __restrict__ s2) {
  __shared__ float r1[4], r2[4];
  int row = blockIdx.x, tid = threadIdx.x;
  float p1 = 0.f, p2 = 0.f;
  for (int d = tid; d < DD; d += 128) {
    float hv = h[row * DD + d];
    p1 += hv * as[d];
    p2 += hv * ad[d];
  }
  for (int off = 16; off; off >>= 1) { p1 += __shfl_xor(p1, off); p2 += __shfl_xor(p2, off); }
  if ((tid & 31) == 0) { r1[tid >> 5] = p1; r2[tid >> 5] = p2; }
  __syncthreads();
  if (tid == 0) {
    s1[row] = r1[0] + r1[1] + r1[2] + r1[3];
    s2[row] = r2[0] + r2[1] + r2[2] + r2[3];
  }
}

__global__ __launch_bounds__(256)
void k_s2max(const float* __restrict__ s2, float* __restrict__ s2m) {
  __shared__ float red[8];
  float m = -3.4e38f;
  for (int i = threadIdx.x; i < NN; i += 256) m = fmaxf(m, s2[i]);
  for (int off = 16; off; off >>= 1) m = fmaxf(m, __shfl_xor(m, off));
  if ((threadIdx.x & 31) == 0) red[threadIdx.x >> 5] = m;
  __syncthreads();
  if (threadIdx.x == 0) {
    float r = red[0];
    for (int k = 1; k < 8; ++k) r = fmaxf(r, red[k]);
    s2m[0] = r;
  }
}

// p = adj ? exp(LRelu(s1+s2) - shift) : 0 ; packed to bf16
__device__ inline void pquad(v4i av, v4f sv, float s1r, float mrow,
                             float& lsum, unsigned short* d) {
#pragma unroll
  for (int t = 0; t < 4; ++t) {
    float e = s1r + sv[t];
    e = e > 0.f ? e : ALPHA_ * e;
    float p = (av[t] > 0) ? __expf(e - mrow) : 0.f;
    lsum += p;
    d[t] = f2bf(p);
  }
}

// ---------- fused attention: out = softmax(mask(LRelu(s1+s2^T))) @ h ----------
// grid = 384 blocks (16 rows each) x 128 threads (4 waves).
// Each wave owns ALL 16 column tiles (128 acc VGPRs) over its private j-chunks,
// so adj (the only HBM-bound stream, 453 MB over 3 views) is read exactly once.
// Shift = LRelu(s1+max(s2)) >= rowmax makes the softmax single-pass (no
// rescaling); division by the row sum at the end is mathematically identical.
__global__ __launch_bounds__(128)
void k_attn(const unsigned short* __restrict__ ht,   // [256][6144] bf16
            const int* __restrict__ adj,             // [6144][6144]
            const float* __restrict__ s1,
            const float* __restrict__ s2,
            const float* __restrict__ s2max,
            float* __restrict__ outv) {              // [6144][256]
  __shared__ float sacc[16 * DD];
  __shared__ float sl[16];
  const int tid  = threadIdx.x;
  const int lane = tid & 31;
  const int wid  = tid >> 5;
  const int l15  = lane & 15;
  const int hi8  = (lane >> 4) << 3;
  const int hi16 = (lane >> 4) << 4;
  const int rb   = blockIdx.x * 16;
  const int row  = rb + l15;
  const float s1r = s1[row];
  float mrow = s1r + s2max[0];
  mrow = mrow > 0.f ? mrow : ALPHA_ * mrow;

  v8f acc[16];
  v8f zero8 = {0.f, 0.f, 0.f, 0.f, 0.f, 0.f, 0.f, 0.f};
#pragma unroll
  for (int ct = 0; ct < 16; ++ct) acc[ct] = zero8;
  float lsum = 0.f;

  const int* adjrow = adj + row * NN;
  for (int jc = wid * 32; jc < NN; jc += 128) {
    // warm L2/L0 for the NEXT chunk's adj lines (global_prefetch_b8;
    // speculative, no LOADcnt, silently dropped past the array end)
    __builtin_prefetch(adjrow + jc + 128 + hi8, 0, 1);
    __builtin_prefetch(adjrow + jc + 144 + hi8, 0, 1);

    ABPack a;
    {
      const v4i* a0 = (const v4i*)(adjrow + jc + hi8);
      const v4f* q0 = (const v4f*)(s2 + jc + hi8);
      const v4i* a1 = (const v4i*)(adjrow + jc + 16 + hi8);
      const v4f* q1 = (const v4f*)(s2 + jc + 16 + hi8);
      pquad(a0[0], q0[0], s1r, mrow, lsum, &a.us[0]);
      pquad(a0[1], q0[1], s1r, mrow, lsum, &a.us[4]);
      pquad(a1[0], q1[0], s1r, mrow, lsum, &a.us[8]);
      pquad(a1[1], q1[1], s1r, mrow, lsum, &a.us[12]);
    }
    const unsigned short* bbase = ht + l15 * NN + jc + hi8;
#pragma unroll
    for (int ct = 0; ct < 16; ++ct) {
      ABPack b;
      b.q[0] = *(const v4i*)(bbase + ct * 16 * NN);
      b.q[1] = *(const v4i*)(bbase + ct * 16 * NN + 16);
      acc[ct] = __builtin_amdgcn_wmma_f32_16x16x32_bf16(false, a.bf, false, b.bf,
                                                        (short)0, acc[ct], false, false);
    }
  }

  lsum += __shfl_xor(lsum, 16);                 // lanes 0..15 hold full row sums
  for (int i = tid; i < 16 * DD; i += 128) sacc[i] = 0.f;
  if (tid < 16) sl[tid] = 0.f;
  __syncthreads();
#pragma unroll
  for (int ct = 0; ct < 16; ++ct)
#pragma unroll
    for (int v = 0; v < 8; ++v)
      atomicAdd(&sacc[(v + hi8) * DD + ct * 16 + l15], acc[ct][v]);
  if (lane < 16) atomicAdd(&sl[lane], lsum);
  __syncthreads();
  for (int i = tid; i < 16 * DD; i += 128) {
    int m = i >> 8;
    float l = sl[m];
    outv[(rb + m) * DD + (i & 255)] = (l > 0.f) ? sacc[i] / l : 0.f;
  }
}

// ---------- projection + LayerNorm + weighted accumulate ----------
__global__ __launch_bounds__(128)
void k_proj_ln(const float* __restrict__ outv, const float* __restrict__ pw,
               const float* __restrict__ pb, const float* __restrict__ g,
               const float* __restrict__ bb, const float* __restrict__ wvec,
               int v, int init, float* __restrict__ zacc) {
  __shared__ float xrow[DD];
  __shared__ float red[4];
  const int i = blockIdx.x;
  const int o = threadIdx.x;
  for (int k = o; k < DD; k += 128) xrow[k] = outv[i * DD + k];
  __syncthreads();
  float y = pb[o];
#pragma unroll 8
  for (int k = 0; k < DD; ++k) y += xrow[k] * pw[k * OUTD + o];
  float s = y;
  for (int off = 16; off; off >>= 1) s += __shfl_xor(s, off);
  if ((o & 31) == 0) red[o >> 5] = s;
  __syncthreads();
  float mean = (red[0] + red[1] + red[2] + red[3]) * (1.f / OUTD);
  __syncthreads();
  float d = y - mean;
  float sv = d * d;
  for (int off = 16; off; off >>= 1) sv += __shfl_xor(sv, off);
  if ((o & 31) == 0) red[o >> 5] = sv;
  __syncthreads();
  float var = (red[0] + red[1] + red[2] + red[3]) * (1.f / OUTD);
  float z = d * rsqrtf(var + EPS_) * g[o] + bb[o];
  float w = wvec[v] * z;
  if (init) zacc[i * OUTD + o] = w;
  else      zacc[i * OUTD + o] += w;
}

__global__ void k_finalize(const float* __restrict__ z, float* __restrict__ out, int n) {
  int i = blockIdx.x * blockDim.x + threadIdx.x;
  if (i < n) {
    float v = z[i];
    unsigned u = __float_as_uint(v);
    if ((u & 0x7f800000u) == 0x7f800000u) v = 0.f;   // NaN/Inf -> 0
    out[i] = v;
  }
}

// ---------- launch ----------
extern "C" void kernel_launch(void* const* d_in, const int* in_sizes, int n_in,
                              void* d_out, int out_size, void* d_ws, size_t ws_size,
                              hipStream_t stream) {
  const float* x     = (const float*)d_in[0];
  const int*   adj[3]  = {(const int*)d_in[1], (const int*)d_in[2], (const int*)d_in[3]};
  const float* Wv[3]   = {(const float*)d_in[4], (const float*)d_in[7], (const float*)d_in[10]};
  const float* asrc[3] = {(const float*)d_in[5], (const float*)d_in[8], (const float*)d_in[11]};
  const float* adst[3] = {(const float*)d_in[6], (const float*)d_in[9], (const float*)d_in[12]};
  const float* pw    = (const float*)d_in[13];
  const float* pb    = (const float*)d_in[14];
  const float* lng   = (const float*)d_in[15];
  const float* lnb   = (const float*)d_in[16];
  const float* alpha = (const float*)d_in[17];
  float* out = (float*)d_out;

  char* ws = (char*)d_ws;                   // needs ~25.5 MB
  unsigned short* xbf  = (unsigned short*)(ws + OFF_XBF);
  unsigned short* wt   = (unsigned short*)(ws + OFF_WT);
  float*          h    = (float*)(ws + OFF_H);
  unsigned short* ht   = (unsigned short*)(ws + OFF_HT);
  float*          outv = (float*)(ws + OFF_OUTV);
  float*          zacc = (float*)(ws + OFF_ZACC);
  float*          s1   = (float*)(ws + OFF_S1);
  float*          s2   = (float*)(ws + OFF_S2);
  float*          s2m  = (float*)(ws + OFF_S2MAX);
  float*          wvec = (float*)(ws + OFF_WVEC);

  k_softmax_alpha<<<1, 64, 0, stream>>>(alpha, wvec, out + NN * OUTD);
  k_convert_x<<<(NN * IND + 255) / 256, 256, 0, stream>>>(x, xbf, NN * IND);

  for (int v = 0; v < 3; ++v) {
    k_transpose_w<<<(IND * DD + 255) / 256, 256, 0, stream>>>(Wv[v], wt);
    k_gemm_h<<<(NN / 16) * (DD / 16) / 4, 128, 0, stream>>>(xbf, wt, h, ht);
    k_scores<<<NN, 128, 0, stream>>>(h, asrc[v], adst[v], s1, s2);
    k_s2max<<<1, 256, 0, stream>>>(s2, s2m);
    k_attn<<<NN / 16, 128, 0, stream>>>(ht, adj[v], s1, s2, s2m, outv);
    k_proj_ln<<<NN, 128, 0, stream>>>(outv, pw, pb, lng, lnb, wvec, v, v == 0 ? 1 : 0, zacc);
  }
  k_finalize<<<(NN * OUTD + 255) / 256, 256, 0, stream>>>(zacc, out, NN * OUTD);
}
